// GATLayer_32985348833757
// MI455X (gfx1250) — compile-verified, matrix operand order checked by reading, not run on hardware
//
#include <hip/hip_runtime.h>
#include <hip/hip_bf16.h>

typedef __attribute__((ext_vector_type(16))) _Float16 v16h;
typedef __attribute__((ext_vector_type(8)))  _Float16 v8h;
typedef __attribute__((ext_vector_type(8)))  float    v8f;

static constexpr int   N_    = 50000;
static constexpr int   E_    = 800000;
static constexpr int   INF_  = 128;
static constexpr int   OUTF_ = 64;
static constexpr int   H_    = 4;
static constexpr int   COLS_ = H_ * OUTF_;   // 256
static constexpr float SLOPE_ = 0.2f;
static constexpr float EPS_   = 1e-8f;

// ---------------------------------------------------------------- utilities
__global__ void zero_f32(float* p, int n) {
    int i = blockIdx.x * blockDim.x + threadIdx.x;
    if (i < n) p[i] = 0.0f;
}

__global__ void cvt_x_f16(const float* __restrict__ x, _Float16* __restrict__ xh, int n) {
    int i = blockIdx.x * blockDim.x + threadIdx.x;
    if (i < n) xh[i] = (_Float16)x[i];
}

// Pack W (H,128,64) f32 into the wave32 WMMA B-fragment layout:
// Bpack[t(16 col-tiles)][ks(4 K-steps)][lane(32)][elem(16 f16)]
// per-lane element i maps to K = ks*32 + (i<8 ? k0+i : k0+8+i), k0 = (lane>=16)?8:0,
// column n = t*16 + (lane&15)  (mirrors the 16-bit A-matrix K striping).
__global__ void pack_w(const float* __restrict__ W, _Float16* __restrict__ Bpack) {
    int id = blockIdx.x * blockDim.x + threadIdx.x;   // 16*4*32*16 = 32768
    if (id >= 16 * 4 * 32 * 16) return;
    int i  =  id        & 15;
    int L  = (id >> 4)  & 31;
    int ks = (id >> 9)  & 3;
    int t  =  id >> 11;
    int k0 = (L >> 4) << 3;
    int kk = (i < 8) ? (k0 + i) : (k0 + 8 + i);
    int K  = ks * 32 + kk;
    int col  = t * 16 + (L & 15);
    int head = col >> 6;
    int f    = col & 63;
    Bpack[id] = (_Float16)W[(head * INF_ + K) * OUTF_ + f];
}

// ---------------------------------------------------------------- WMMA GEMM
// Wh(N x 256) = xh(N x 128, f16) * Wpack(128 x 256), f32 accumulate.
// One block = 8 waves; block handles 16 rows; wave w handles col tiles 2w, 2w+1.
union AFrag { v16h v; struct { v8h lo, hi; } p; };

__global__ __launch_bounds__(256)
void gemm_wmma(const _Float16* __restrict__ xh,
               const _Float16* __restrict__ Bpack,
               float* __restrict__ Wh) {
    const int wave = threadIdx.x >> 5;
    const int lane = threadIdx.x & 31;
    const int m0   = blockIdx.x * 16;        // N = 3125 * 16 exactly
    const int mrow = lane & 15;
    const int k0   = (lane >> 4) << 3;       // 0 or 8

    const _Float16* xrow = xh + (m0 + mrow) * INF_ + k0;
    const int t0 = wave * 2;
    const int t1 = t0 + 1;

    v8f c0 = {};
    v8f c1 = {};

#pragma unroll
    for (int ks = 0; ks < 4; ++ks) {
        AFrag a;
        a.p.lo = *(const v8h*)(xrow + ks * 32);
        a.p.hi = *(const v8h*)(xrow + ks * 32 + 16);
        v16h b0 = *(const v16h*)(Bpack + (((t0 * 4 + ks) * 32) + lane) * 16);
        v16h b1 = *(const v16h*)(Bpack + (((t1 * 4 + ks) * 32) + lane) * 16);
        c0 = __builtin_amdgcn_wmma_f32_16x16x32_f16(false, a.v, false, b0,
                                                    (short)0, c0, false, false);
        c1 = __builtin_amdgcn_wmma_f32_16x16x32_f16(false, a.v, false, b1,
                                                    (short)0, c1, false, false);
    }

    // D layout: VGPR r -> row m0 + r + (lane>=16 ? 8 : 0), col = tile*16 + (lane&15)
    const int rowOff = (lane >> 4) << 3;
    const int col0   = t0 * 16 + (lane & 15);
    const int col1   = col0 + 16;
#pragma unroll
    for (int r = 0; r < 8; ++r) {
        float* dstRow = Wh + (size_t)(m0 + rowOff + r) * COLS_;
        dstRow[col0] = c0[r];
        dstRow[col1] = c1[r];
    }
}

// ---------------------------------------------------------------- attention
// e_src[h][n] = dot(Wh[n, h*64 : h*64+64], a_src[h]); likewise e_dst.
__global__ void node_scores(const float* __restrict__ Wh,
                            const float* __restrict__ a_src,
                            const float* __restrict__ a_dst,
                            float* __restrict__ es,
                            float* __restrict__ ed) {
    int tid = blockIdx.x * blockDim.x + threadIdx.x;
    if (tid >= N_ * H_) return;
    int h = tid & (H_ - 1);
    int n = tid >> 2;
    const float* w  = Wh + (size_t)n * COLS_ + h * OUTF_;
    const float* as = a_src + h * OUTF_;
    const float* ad = a_dst + h * OUTF_;
    float ss = 0.0f, sd = 0.0f;
#pragma unroll
    for (int f = 0; f < OUTF_; f += 4) {
        float4 wv = *(const float4*)(w + f);
        float4 av = *(const float4*)(as + f);
        float4 dv = *(const float4*)(ad + f);
        ss += wv.x * av.x + wv.y * av.y + wv.z * av.z + wv.w * av.w;
        sd += wv.x * dv.x + wv.y * dv.y + wv.z * dv.z + wv.w * dv.w;
    }
    es[h * N_ + n] = ss;
    ed[h * N_ + n] = sd;
}

// Per-edge logits + segment max (effective max is max(seg_max, 0): segmax
// initialized to 0.0f bits; only positive values raise it, and positive-float
// bit patterns are monotone under integer max).
__global__ void edge_logits(const float* __restrict__ es,
                            const float* __restrict__ ed,
                            const long long* __restrict__ ei,
                            float* __restrict__ ebuf,
                            int* __restrict__ segmax) {
    int e = blockIdx.x * blockDim.x + threadIdx.x;
    if (e >= E_) return;
    int s = (int)ei[e];
    int d = (int)ei[E_ + e];
#pragma unroll
    for (int h = 0; h < H_; ++h) {
        float v = es[h * N_ + s] + ed[h * N_ + d];
        v = (v > 0.0f) ? v : SLOPE_ * v;
        ebuf[h * E_ + e] = v;
        if (v > 0.0f) atomicMax(&segmax[h * N_ + d], __float_as_int(v));
    }
}

__global__ void edge_exp(const long long* __restrict__ ei,
                         const int* __restrict__ segmax,
                         float* __restrict__ ebuf,
                         float* __restrict__ expsum) {
    int e = blockIdx.x * blockDim.x + threadIdx.x;
    if (e >= E_) return;
    int d = (int)ei[E_ + e];
#pragma unroll
    for (int h = 0; h < H_; ++h) {
        float m  = __int_as_float(segmax[h * N_ + d]);   // = max(seg_max, 0)
        float ee = __expf(ebuf[h * E_ + e] - m);
        ebuf[h * E_ + e] = ee;
        atomicAdd(&expsum[h * N_ + d], ee);
    }
}

// out[dst, h*64+f] += att * Wh[src, h*64+f]; one edge handled per 256-thread
// block iteration, grid-strided, with a prefetch of the next gather row.
__global__ __launch_bounds__(256)
void aggregate(const float* __restrict__ Wh,
               const float* __restrict__ ebuf,
               const float* __restrict__ expsum,
               const long long* __restrict__ ei,
               float* __restrict__ out) {
    const int t = threadIdx.x;          // 0..255 -> (h, f)
    const int h = t >> 6;
    for (int e = blockIdx.x; e < E_; e += gridDim.x) {
        int s = (int)ei[e];
        int d = (int)ei[E_ + e];
        int en = e + gridDim.x;
        if (en < E_) {
            int sn = (int)ei[en];
            __builtin_prefetch(Wh + (size_t)sn * COLS_ + t, 0, 1);
        }
        float att = ebuf[h * E_ + e] / (expsum[h * N_ + d] + EPS_);
        atomicAdd(&out[(size_t)d * COLS_ + t], att * Wh[(size_t)s * COLS_ + t]);
    }
}

// ---------------------------------------------------------------- launcher
extern "C" void kernel_launch(void* const* d_in, const int* in_sizes, int n_in,
                              void* d_out, int out_size, void* d_ws, size_t ws_size,
                              hipStream_t stream) {
    const float*     x     = (const float*)d_in[0];
    const long long* ei    = (const long long*)d_in[1];   // int64 (2, E)
    const float*     W     = (const float*)d_in[2];
    const float*     a_src = (const float*)d_in[3];
    const float*     a_dst = (const float*)d_in[4];
    float*           out   = (float*)d_out;               // (N, 256)

    // workspace layout (bytes)
    char* ws = (char*)d_ws;
    float*    Wh     = (float*)(ws);                                  // N*256 f32
    _Float16* xh     = (_Float16*)(ws + (size_t)N_ * COLS_ * 4);      // N*128 f16
    _Float16* Bpack  = (_Float16*)(ws + 64000000ull);                 // 32768 f16
    float*    es     = (float*)(ws + 64065536ull);                    // H*N
    float*    ed     = (float*)(ws + 64865536ull);                    // H*N
    float*    segmax = (float*)(ws + 65665536ull);                    // H*N
    float*    expsum = (float*)(ws + 66465536ull);                    // H*N
    float*    ebuf   = (float*)(ws + 67265536ull);                    // H*E

    const int B = 256;

    // zero output and the two reduction buffers (segmax/expsum are adjacent)
    zero_f32<<<(N_ * COLS_ + B - 1) / B, B, 0, stream>>>(out, N_ * COLS_);
    zero_f32<<<(2 * H_ * N_ + B - 1) / B, B, 0, stream>>>(segmax, 2 * H_ * N_);

    // precision staging + weight packing for WMMA
    cvt_x_f16<<<(N_ * INF_ + B - 1) / B, B, 0, stream>>>(x, xh, N_ * INF_);
    pack_w<<<(16 * 4 * 32 * 16 + B - 1) / B, B, 0, stream>>>(W, Bpack);

    // projection GEMM: 3125 row-tiles, 8 waves/block
    gemm_wmma<<<N_ / 16, 256, 0, stream>>>(xh, Bpack, Wh);

    // attention logits per node, per edge, softmax stats, aggregation
    node_scores<<<(N_ * H_ + B - 1) / B, B, 0, stream>>>(Wh, a_src, a_dst, es, ed);
    edge_logits<<<(E_ + B - 1) / B, B, 0, stream>>>(es, ed, ei, ebuf, (int*)segmax);
    edge_exp<<<(E_ + B - 1) / B, B, 0, stream>>>(ei, (const int*)segmax, ebuf, expsum);
    aggregate<<<8192, 256, 0, stream>>>(Wh, ebuf, expsum, ei, out);
}
